// QKVParallelLinearWithToppings_21036749816231
// MI455X (gfx1250) — matrix operand order repaired and checked by reading, not compiled, not run on hardware
//
#include <hip/hip_runtime.h>
#include <cstdint>
#include <cstddef>

typedef __attribute__((ext_vector_type(16))) __bf16 v16bf;
typedef __attribute__((ext_vector_type(8)))  float  v8f;
typedef __attribute__((ext_vector_type(4)))  __bf16 v4bf;

// ---------------- problem sizes ----------------
constexpr int T    = 4096;
constexpr int DIN  = 2048;
constexpr int DOUT = 6144;     // DQ + 2*DKV
constexpr int NA   = 4;

// ---------------- tiling ----------------
constexpr int TM = 128, TN = 128, TK = 32;
constexpr int ASTR = 40;                  // LDS row stride in halfs (32 + 8 pad, 80B)
constexpr int PERM_LEN = T + NA * TM;     // 4608 (each adapter segment padded to 128)
constexpr int MTILES = PERM_LEN / TM;     // 36
constexpr int NTILES = DOUT / TN;         // 48
constexpr uint32_t BUFBYTES = TM * ASTR * sizeof(__bf16);  // 10240B per LDS buffer

// ---------------- metadata layout (int offsets into ws) ----------------
constexpr int M_COUNT = 0;    // [4]  per-adapter token counts
constexpr int M_CURS  = 4;    // [4]  scatter cursors
constexpr int M_SEG   = 8;    // [4]  padded segment base (rows)
constexpr int M_TILEA = 16;   // [48] tile -> adapter (-1 = dead tile)
constexpr int M_PERM  = 64;   // [4608] grouped token permutation (-1 = pad row)
constexpr int M_TOTAL = M_PERM + PERM_LEN;   // 4672 ints

// ============================================================
// metadata kernels
// ============================================================
__global__ void init_meta(int* meta) {
  int i = blockIdx.x * blockDim.x + threadIdx.x;
  if (i >= M_TOTAL) return;
  meta[i] = (i < 16) ? 0 : -1;
}

__global__ void count_tokens(const int* __restrict__ idx, int* meta) {
  int t = blockIdx.x * blockDim.x + threadIdx.x;
  if (t < T) atomicAdd(&meta[M_COUNT + idx[t]], 1);
}

__global__ void plan_tiles(int* meta) {
  if (blockIdx.x == 0 && threadIdx.x == 0) {
    int off = 0;
    for (int a = 0; a < NA; ++a) {
      meta[M_SEG + a] = off;
      int c  = meta[M_COUNT + a];
      int nt = (c + TM - 1) / TM;
      for (int i = 0; i < nt; ++i) meta[M_TILEA + off / TM + i] = a;
      off += nt * TM;
    }
  }
}

__global__ void scatter_tokens(const int* __restrict__ idx, int* meta) {
  int t = blockIdx.x * blockDim.x + threadIdx.x;
  if (t < T) {
    int a = idx[t];
    int p = atomicAdd(&meta[M_CURS + a], 1);
    meta[M_PERM + meta[M_SEG + a] + p] = t;
  }
}

// ============================================================
// x -> bf16
// ============================================================
__global__ void convert_x(const float* __restrict__ x, __bf16* __restrict__ xb) {
  int i = blockIdx.x * blockDim.x + threadIdx.x;
  if (i < T * DIN / 4) {
    float4 v = ((const float4*)x)[i];
    v4bf o;
    o.x = (__bf16)v.x; o.y = (__bf16)v.y; o.z = (__bf16)v.z; o.w = (__bf16)v.w;
    ((v4bf*)xb)[i] = o;
  }
}

// ============================================================
// Wcomb[a][n][k] = W_base[n][k] + Delta[a][k][n] + sum_r A[a][k][r]*B[a][r][n]
// 64(k) x 64(n) tiles; Delta transposed through LDS; LoRA folded (rank 16).
// ============================================================
__global__ __launch_bounds__(256) void fold_weights(
    const float* __restrict__ Wb,    // [DOUT][DIN]
    const float* __restrict__ Abuf,  // [NA][DIN][48]
    const float* __restrict__ Bq,    // [NA][16][2048]
    const float* __restrict__ Bkv,   // [NA][16][4096]
    const float* __restrict__ Dq,    // [NA][DIN][2048]
    const float* __restrict__ Dkv,   // [NA][DIN][4096]
    __bf16* __restrict__ wcomb)      // [NA][DOUT][DIN]
{
  __shared__ float dT[64 * 65];   // delta tile, [k][n], padded
  __shared__ float As[64 * 17];   // A tile, [k][r], padded
  __shared__ float Bsh[16 * 64];  // B tile, [r][n]

  const int nbase = blockIdx.x * 64;
  const int kbase = blockIdx.y * 64;
  const int a     = blockIdx.z;
  const int s     = nbase >> 11;             // 0=q, 1=k, 2=v (tile never straddles)
  const int rsel  = s * 16;

  const float* dsrc; const float* bsrc; int ncols, ncoff;
  if (s == 0) { dsrc = Dq  + (size_t)a * DIN * 2048; bsrc = Bq  + (size_t)a * 16 * 2048; ncols = 2048; ncoff = nbase; }
  else        { dsrc = Dkv + (size_t)a * DIN * 4096; bsrc = Bkv + (size_t)a * 16 * 4096; ncols = 4096; ncoff = nbase - 2048; }

  const int tid = threadIdx.x;

  #pragma unroll
  for (int i = 0; i < 16; ++i) {             // 64x64 delta tile
    int idx = tid + i * 256, kk = idx >> 6, nn = idx & 63;
    dT[kk * 65 + nn] = dsrc[(size_t)(kbase + kk) * ncols + ncoff + nn];
  }
  #pragma unroll
  for (int i = 0; i < 4; ++i) {              // 64x16 A tile
    int idx = tid + i * 256, kk = idx >> 4, rr = idx & 15;
    As[kk * 17 + rr] = Abuf[((size_t)a * DIN + kbase + kk) * 48 + rsel + rr];
  }
  #pragma unroll
  for (int i = 0; i < 4; ++i) {              // 16x64 B tile
    int idx = tid + i * 256, rr = idx >> 6, nn = idx & 63;
    Bsh[rr * 64 + nn] = bsrc[(size_t)rr * ncols + ncoff + nn];
  }
  __syncthreads();

  #pragma unroll
  for (int i = 0; i < 16; ++i) {
    int idx = tid + i * 256, nl = idx >> 6, kl = idx & 63;
    float acc = Wb[(size_t)(nbase + nl) * DIN + kbase + kl] + dT[kl * 65 + nl];
    #pragma unroll
    for (int r = 0; r < 16; ++r) acc += As[kl * 17 + r] * Bsh[r * 64 + nl];
    wcomb[((size_t)a * DOUT + nbase + nl) * DIN + kbase + kl] = (__bf16)acc;
  }
}

// ============================================================
// CDNA5 async memory->LDS staging (ASYNCcnt path, ISA §10.7/§15.18.3)
// ============================================================
__device__ __forceinline__ void async_ld_b128(uint32_t lds_byte_off, uint64_t gaddr) {
  // GV mode: 64-bit global address in VGPR pair, LDS byte address in VGPR.
  asm volatile("global_load_async_to_lds_b128 %0, %1, off"
               :: "v"(lds_byte_off), "v"(gaddr) : "memory");
}
__device__ __forceinline__ void wait_asynccnt0() {
  asm volatile("s_wait_asynccnt 0x0" ::: "memory");
}

// ============================================================
// Main GEMM: out[perm-rows, :] = xb[perm-rows, :] @ Wcomb[a]
// 128x128 tile, K-step 32, 8 waves (2M x 4N), double-buffered LDS filled
// by global_load_async_to_lds_b128; v_wmma_f32_16x16x32_bf16 fragments
// per ISA 7.12.2 layouts.
// ============================================================
union Frag { uint4 q[2]; v16bf v; };

__global__ __launch_bounds__(256) void qkv_gemm(
    const __bf16* __restrict__ xb,      // [T][DIN] bf16
    const __bf16* __restrict__ wcomb,   // [NA][DOUT][DIN] bf16
    const int*    __restrict__ meta,
    float*        __restrict__ out)     // [T][DOUT] f32
{
  __shared__ __align__(16) __bf16 AsT[2][TM * ASTR];
  __shared__ __align__(16) __bf16 BsT[2][TN * ASTR];
  __shared__ int permS[TM];

  const int mtile = blockIdx.y;
  const int ntile = blockIdx.x;
  const int adap  = meta[M_TILEA + mtile];
  if (adap < 0) return;                       // fully-padded tile

  const int tid  = threadIdx.x;
  const int lane = tid & 31;
  const int wave = tid >> 5;
  const int wm   = wave & 1;                  // 2 wave rows  (64 M each)
  const int wn   = wave >> 1;                 // 4 wave cols  (32 N each)

  const int* perm = meta + M_PERM + mtile * TM;
  if (tid < TM) permS[tid] = perm[tid];

  // cooperative staging: thread owns 8 halfs (16B) of rows lr0 and lr0+64
  const int lr0 = tid >> 2;
  const int lr1 = lr0 + 64;
  const int seg = (tid & 3) * 8;              // half offset 0/8/16/24

  int pr0 = perm[lr0]; if (pr0 < 0) pr0 = 0;  // pad rows compute garbage, never stored
  int pr1 = perm[lr1]; if (pr1 < 0) pr1 = 0;

  const uint64_t xr0 = (uint64_t)(uintptr_t)(xb + (size_t)pr0 * DIN + seg);
  const uint64_t xr1 = (uint64_t)(uintptr_t)(xb + (size_t)pr1 * DIN + seg);
  const __bf16* wb0  = wcomb + ((size_t)adap * DOUT + (size_t)ntile * TN) * DIN + seg;
  const uint64_t wr0 = (uint64_t)(uintptr_t)(wb0 + (size_t)lr0 * DIN);
  const uint64_t wr1 = (uint64_t)(uintptr_t)(wb0 + (size_t)lr1 * DIN);

  // LDS byte offsets (low 32 bits of generic shared pointer = LDS address)
  const uint32_t ldsA0 = (uint32_t)(uintptr_t)&AsT[0][lr0 * ASTR + seg];
  const uint32_t ldsA1 = (uint32_t)(uintptr_t)&AsT[0][lr1 * ASTR + seg];
  const uint32_t ldsB0 = (uint32_t)(uintptr_t)&BsT[0][lr0 * ASTR + seg];
  const uint32_t ldsB1 = (uint32_t)(uintptr_t)&BsT[0][lr1 * ASTR + seg];

  v8f acc[4][2];
  #pragma unroll
  for (int m = 0; m < 4; ++m)
    #pragma unroll
    for (int n = 0; n < 2; ++n) acc[m][n] = (v8f)0.0f;

  // prologue: async-stage k-tile 0 into buffer 0
  async_ld_b128(ldsA0, xr0);
  async_ld_b128(ldsA1, xr1);
  async_ld_b128(ldsB0, wr0);
  async_ld_b128(ldsB1, wr1);
  wait_asynccnt0();
  __syncthreads();

  // fragment addressing (ISA 7.12.2, 16-bit A 16x32 / B 32x16 layouts)
  const int frow = lane & 15;
  const int akh  = (lane >> 4) * 8;    // A: K {0..7,16..23} | {8..15,24..31}
  const int bkh  = (lane >> 4) * 8;    // B: K {0..7,16..23} | {8..15,24..31}

  constexpr int KSTEPS = DIN / TK;     // 64
  for (int kt = 0; kt < KSTEPS; ++kt) {
    const int cur = kt & 1;
    if (kt + 1 < KSTEPS) {
      const uint32_t bo = ((kt + 1) & 1) ? BUFBYTES : 0;   // next LDS buffer
      const uint64_t ko = (uint64_t)(kt + 1) * (TK * sizeof(__bf16));
      async_ld_b128(ldsA0 + bo, xr0 + ko);
      async_ld_b128(ldsA1 + bo, xr1 + ko);
      async_ld_b128(ldsB0 + bo, wr0 + ko);
      async_ld_b128(ldsB1 + bo, wr1 + ko);
    }

    Frag af[4], bf2[2];
    #pragma unroll
    for (int m = 0; m < 4; ++m) {
      const int r = wm * 64 + m * 16 + frow;
      af[m].q[0] = *(const uint4*)&AsT[cur][r * ASTR + akh];
      af[m].q[1] = *(const uint4*)&AsT[cur][r * ASTR + akh + 16];
    }
    #pragma unroll
    for (int n = 0; n < 2; ++n) {
      const int r = wn * 32 + n * 16 + frow;
      bf2[n].q[0] = *(const uint4*)&BsT[cur][r * ASTR + bkh];
      bf2[n].q[1] = *(const uint4*)&BsT[cur][r * ASTR + bkh + 16];
    }

    #pragma unroll
    for (int m = 0; m < 4; ++m)
      #pragma unroll
      for (int n = 0; n < 2; ++n)
        acc[m][n] = __builtin_amdgcn_wmma_f32_16x16x32_bf16(
            false, af[m].v, false, bf2[n].v, (short)0, acc[m][n], false, false);

    if (kt + 1 < KSTEPS) wait_asynccnt0();   // my async writes to next buffer done
    __syncthreads();                         // everyone's writes visible / reads done
  }

  // epilogue: C/D layout — lane<16: N=lane, M=v; lane>=16: N=lane-16, M=8+v
  const int ncol0 = ntile * TN + wn * 32 + (lane & 15);
  const int mb    = wm * 64 + (lane >> 4) * 8;
  #pragma unroll
  for (int m = 0; m < 4; ++m) {
    #pragma unroll
    for (int n = 0; n < 2; ++n) {
      const int ncol = ncol0 + n * 16;
      #pragma unroll
      for (int v = 0; v < 8; ++v) {
        const int row = permS[mb + m * 16 + v];
        if (row >= 0) out[(size_t)row * DOUT + ncol] = acc[m][n][v];
      }
    }
  }
}

// ============================================================
// launch
// ============================================================
extern "C" void kernel_launch(void* const* d_in, const int* in_sizes, int n_in,
                              void* d_out, int out_size, void* d_ws, size_t ws_size,
                              hipStream_t stream) {
  const float* x    = (const float*)d_in[0];
  const float* Wb   = (const float*)d_in[1];
  const float* Abuf = (const float*)d_in[2];
  const float* Bq   = (const float*)d_in[3];
  const float* Bkv  = (const float*)d_in[4];
  const float* Dq   = (const float*)d_in[5];
  const float* Dkv  = (const float*)d_in[6];
  const int*   idx  = (const int*)d_in[7];
  float* out = (float*)d_out;

  // workspace carve-up: [0,32K) metadata | x_bf16 (16MB) | Wcomb bf16 (96MB)
  int*    meta  = (int*)d_ws;
  __bf16* xb    = (__bf16*)((char*)d_ws + 32768);
  __bf16* wcomb = (__bf16*)((char*)d_ws + 32768 + (size_t)T * DIN * sizeof(__bf16));

  init_meta<<<(M_TOTAL + 255) / 256, 256, 0, stream>>>(meta);
  count_tokens<<<T / 256, 256, 0, stream>>>(idx, meta);
  plan_tiles<<<1, 1, 0, stream>>>(meta);
  scatter_tokens<<<T / 256, 256, 0, stream>>>(idx, meta);
  convert_x<<<(T * DIN / 4 + 255) / 256, 256, 0, stream>>>(x, xb);
  fold_weights<<<dim3(DOUT / 64, DIN / 64, NA), 256, 0, stream>>>(Wb, Abuf, Bq, Bkv, Dq, Dkv, wcomb);
  qkv_gemm<<<dim3(NTILES, MTILES), 256, 0, stream>>>(xb, wcomb, meta, out);
}
